// MultiHeadSelfAttention_12773232738393
// MI455X (gfx1250) — matrix operand order repaired and checked by reading, not compiled, not run on hardware
//
#include <hip/hip_runtime.h>
#include <hip/hip_bf16.h>

// ---------------------------------------------------------------------------
// Causal multi-head self-attention for MI455X (gfx1250, wave32, WMMA).
//   B = S = 256, E = 768, H = 12, d = 64
// Pipeline:
//   1) cvt X f32->f16                 (Xh   : 96 MB)
//   2) pack W{q,k,v} -> Wh f16 [2304x768], bias concat (3.5 MB)
//   3) QKV = Xh @ Wh^T + b  via v_wmma_f32_16x16x32_f16
//        -> Qb[bh][s][d], Kb[bh][s][d], Vt[bh][d][s]   (f16, ~302 MB)
//   4) fused causal attention per (b,h,qblock64), scores in 64KB LDS,
//      f32 softmax, P f16 in-place, PV via WMMA, f32 out.
// Workspace requirement: ~406 MB.
// ---------------------------------------------------------------------------

typedef __attribute__((ext_vector_type(16))) _Float16 v16h;
typedef __attribute__((ext_vector_type(8)))  _Float16 v8h;
typedef __attribute__((ext_vector_type(4)))  _Float16 v4h;
typedef __attribute__((ext_vector_type(8)))  float    v8f;

#define EMB   768
#define NTOT  2304   // 3*EMB (Q,K,V packed)
#define SEQ   256
#define DH    64
#define NHEAD 12

static __device__ __forceinline__ v8f wmma_f16(v16h a, v16h b, v8f c) {
    // D = A(16x32 f16) x B(32x16 f16) + C(16x16 f32)
    return __builtin_amdgcn_wmma_f32_16x16x32_f16(
        /*neg_a=*/false, a, /*neg_b=*/false, b,
        /*c_mod=*/(short)0, c, /*reuse_a=*/false, /*reuse_b=*/false);
}

// A fragment (16x32, f16) from row-major A[ld] at tile (m0,k0).
// CDNA5 layout: lane L -> M = L%16; halves 0..7 = K (o..o+7), 8..15 = K (o+16..o+23), o = 8*(L/16).
static __device__ __forceinline__ v16h load_a_frag(const _Float16* __restrict__ A,
                                                   int ld, int m0, int k0, int lane) {
    const int m  = m0 + (lane & 15);
    const int ko = k0 + ((lane >> 4) << 3);
    const _Float16* p = A + (size_t)m * ld + ko;
    v8h lo = *(const v8h*)(p);
    v8h hi = *(const v8h*)(p + 16);
    v16h f;
#pragma unroll
    for (int i = 0; i < 8; ++i) { f[i] = lo[i]; f[i + 8] = hi[i]; }
    return f;
}

// B fragment (32x16, f16) where logical B[k][n] = W[n][k], W row-major [N][ld].
// CDNA5 layout: lane L -> N = L%16; halves 0..7 = K (o..o+7), 8..15 = K (o+16..o+23), o = 8*(L/16).
static __device__ __forceinline__ v16h load_b_frag(const _Float16* __restrict__ W,
                                                   int ld, int n0, int k0, int lane) {
    const int n  = n0 + (lane & 15);
    const int kb = k0 + ((lane >> 4) << 3);
    const _Float16* p = W + (size_t)n * ld + kb;
    v8h lo = *(const v8h*)(p);
    v8h hi = *(const v8h*)(p + 16);
    v16h f;
#pragma unroll
    for (int i = 0; i < 8; ++i) { f[i] = lo[i]; f[i + 8] = hi[i]; }
    return f;
}

// ---------------------------------------------------------------------------
// 1) f32 -> f16 convert (vectorized x4)
__global__ void attn_cvt_f16_kernel(const float* __restrict__ src,
                                    _Float16* __restrict__ dst, long long n) {
    long long i = ((long long)blockIdx.x * blockDim.x + threadIdx.x) * 4;
    if (i + 3 < n) {
        float4 v = *(const float4*)(src + i);
        v4h h;
        h[0] = (_Float16)v.x; h[1] = (_Float16)v.y;
        h[2] = (_Float16)v.z; h[3] = (_Float16)v.w;
        *(v4h*)(dst + i) = h;
    }
}

// 2) pack weights into Wh[2304][768] f16 and bias concat (f32)
__global__ void attn_pack_w_kernel(const float* __restrict__ Wq,
                                   const float* __restrict__ Wk,
                                   const float* __restrict__ Wv,
                                   const float* __restrict__ bq,
                                   const float* __restrict__ bk,
                                   const float* __restrict__ bv,
                                   _Float16* __restrict__ Wh,
                                   float* __restrict__ bcat) {
    int idx = blockIdx.x * blockDim.x + threadIdx.x;
    if (idx < 3 * EMB * EMB) {
        int which = idx / (EMB * EMB);
        int rem   = idx - which * (EMB * EMB);
        const float* W = (which == 0) ? Wq : ((which == 1) ? Wk : Wv);
        Wh[idx] = (_Float16)W[rem];
        if (idx < NTOT) {
            int wb = idx / EMB, r = idx - wb * EMB;
            const float* bb = (wb == 0) ? bq : ((wb == 1) ? bk : bv);
            bcat[idx] = bb[r];
        }
    }
}

// ---------------------------------------------------------------------------
// 3) QKV projection GEMM: [65536 x 768] x [768 x 2304] + bias
//    block = 256 threads = 8 waves (2 M x 4 N), block tile 128x256,
//    wave tile 64x64 (4x4 fragments), K step 32.
__global__ __launch_bounds__(256) void attn_qkv_proj_kernel(
    const _Float16* __restrict__ Xh,    // [65536][768]
    const _Float16* __restrict__ Wh,    // [2304][768]
    const float*    __restrict__ bcat,  // [2304]
    _Float16* __restrict__ Qb,          // [b*12+h][s][d]
    _Float16* __restrict__ Kb,          // [b*12+h][s][d]
    _Float16* __restrict__ Vt)          // [b*12+h][d][s]
{
    const int lane = threadIdx.x & 31;
    const int w    = threadIdx.x >> 5;       // 0..7
    const int m0   = blockIdx.y * 128 + (w >> 2) * 64;
    const int n0   = blockIdx.x * 256 + (w & 3) * 64;

    v8f acc[4][4];
#pragma unroll
    for (int i = 0; i < 4; ++i)
#pragma unroll
        for (int j = 0; j < 4; ++j) acc[i][j] = (v8f){};

    for (int k0 = 0; k0 < EMB; k0 += 32) {
        if (k0 + 32 < EMB) {
            // warm next A stripe (global_prefetch_b8)
            __builtin_prefetch(Xh + (size_t)(m0 + (lane & 15)) * EMB + k0 + 32, 0, 1);
        }
        v16h a[4], b[4];
#pragma unroll
        for (int i = 0; i < 4; ++i) a[i] = load_a_frag(Xh, EMB, m0 + 16 * i, k0, lane);
#pragma unroll
        for (int j = 0; j < 4; ++j) b[j] = load_b_frag(Wh, EMB, n0 + 16 * j, k0, lane);
#pragma unroll
        for (int i = 0; i < 4; ++i)
#pragma unroll
            for (int j = 0; j < 4; ++j)
                acc[i][j] = wmma_f16(a[i], b[j], acc[i][j]);
    }

    // Epilogue: bias + scatter into attention-friendly layouts.
    // D layout: lane L -> col = n0j + L%16 ; rows = m0i + r + 8*(L/16), r=0..7
    const int nlo = lane & 15;
    const int mhi = (lane >> 4) << 3;
#pragma unroll
    for (int j = 0; j < 4; ++j) {
        const int n    = n0 + 16 * j + nlo;
        const float bv = bcat[n];
        const int kind = n / EMB;             // 0=Q 1=K 2=V (tiles never straddle)
        const int e    = n - kind * EMB;
        const int h    = e >> 6;
        const int dd   = e & 63;
#pragma unroll
        for (int i = 0; i < 4; ++i) {
#pragma unroll
            for (int r = 0; r < 8; ++r) {
                const int m   = m0 + 16 * i + mhi + r;
                const int bb  = m >> 8;       // batch
                const int s   = m & 255;      // sequence pos
                const size_t bh = (size_t)(bb * NHEAD + h);
                const _Float16 hv = (_Float16)(acc[i][j][r] + bv);
                if (kind == 0)      Qb[(bh << 14) + (size_t)s * DH + dd] = hv;
                else if (kind == 1) Kb[(bh << 14) + (size_t)s * DH + dd] = hv;
                else                Vt[(bh << 14) + (size_t)dd * SEQ + s] = hv;
            }
        }
    }
}

// ---------------------------------------------------------------------------
// 4) Fused causal attention. One block per (b,h,qblock64). 256 thr = 8 waves.
//    LDS: 64 rows x 256 f32 scores = 64 KB; reused in-place as f16 P.
__global__ __launch_bounds__(256) void attn_fused_kernel(
    const _Float16* __restrict__ Qb,
    const _Float16* __restrict__ Kb,
    const _Float16* __restrict__ Vt,
    float* __restrict__ Out)            // [256][256][768] f32
{
    __shared__ float sS[64 * SEQ];      // 65536 B

    const int lane   = threadIdx.x & 31;
    const int w      = threadIdx.x >> 5;         // 0..7
    const int qblk   = blockIdx.x & 3;           // q block of 64 rows
    const int bh     = blockIdx.x >> 2;          // 0..3071
    const int q0base = qblk * 64;

    const _Float16* Qp = Qb + ((size_t)bh << 14);
    const _Float16* Kp = Kb + ((size_t)bh << 14);
    const _Float16* Vp = Vt + ((size_t)bh << 14);

    const float scale = 0.125f;                  // 1/sqrt(64)
    const int nlo = lane & 15;
    const int mhi = (lane >> 4) << 3;

    // ---- Phase 1: S = scale * Q Kt with causal mask -> LDS ----
    // 4 q-tiles x 16 k-tiles = 64 tiles; wave w owns tiles [8w, 8w+8)
    for (int t = w * 8; t < w * 8 + 8; ++t) {
        const int qt = t >> 4;
        const int kt = t & 15;
        const int q0 = q0base + qt * 16;
        const int k0 = kt * 16;
        if (k0 > q0 + 15) {                      // tile fully above diagonal
#pragma unroll
            for (int r = 0; r < 8; ++r)
                sS[(qt * 16 + mhi + r) * SEQ + k0 + nlo] = -3.0e38f;
            continue;
        }
        v8f acc = (v8f){};
#pragma unroll
        for (int kk = 0; kk < DH; kk += 32) {
            v16h a = load_a_frag(Qp, DH, q0, kk, lane);
            v16h b = load_b_frag(Kp, DH, k0, kk, lane);   // B[kd][kseq]=K[kseq][kd]
            acc = wmma_f16(a, b, acc);
        }
#pragma unroll
        for (int r = 0; r < 8; ++r) {
            const int q = q0 + mhi + r;
            const int k = k0 + nlo;
            sS[(qt * 16 + mhi + r) * SEQ + k0 + nlo] =
                (k <= q) ? acc[r] * scale : -3.0e38f;
        }
    }
    __syncthreads();

    // ---- Phase 2: row softmax (f32), write P back as f16 in-place ----
    _Float16* sP = (_Float16*)sS;                // P row i at sP + i*512
    for (int rr = 0; rr < 8; ++rr) {
        const int row = w * 8 + rr;              // wave-exclusive row
        float v[8];
#pragma unroll
        for (int i = 0; i < 8; ++i) v[i] = sS[row * SEQ + lane + 32 * i];
        float mx = v[0];
#pragma unroll
        for (int i = 1; i < 8; ++i) mx = fmaxf(mx, v[i]);
        for (int off = 16; off > 0; off >>= 1) mx = fmaxf(mx, __shfl_xor(mx, off, 32));
        float sum = 0.0f;
#pragma unroll
        for (int i = 0; i < 8; ++i) { v[i] = __expf(v[i] - mx); sum += v[i]; }
        for (int off = 16; off > 0; off >>= 1) sum += __shfl_xor(sum, off, 32);
        const float inv = 1.0f / sum;
#pragma unroll
        for (int i = 0; i < 8; ++i)
            sP[row * 512 + lane + 32 * i] = (_Float16)(v[i] * inv);
    }
    __syncthreads();

    // ---- Phase 3: O = P @ V ----
    // 4 q-tiles x 4 d-tiles = 16 output tiles; wave w owns 2.
    const int b = bh / NHEAD;
    const int h = bh - b * NHEAD;
    for (int t = w * 2; t < w * 2 + 2; ++t) {
        const int qt = t >> 2;
        const int dt = t & 3;
        v8f acc = (v8f){};
        for (int k0 = 0; k0 < SEQ; k0 += 32) {
            // A fragment from LDS P (row-major, ld = 512 halves)
            const int m  = qt * 16 + nlo;
            const int ko = k0 + mhi;             // mhi == 8*(lane/16)
            const _Float16* pA = sP + (size_t)m * 512 + ko;
            v8h lo = *(const v8h*)pA;
            v8h hi = *(const v8h*)(pA + 16);
            v16h a;
#pragma unroll
            for (int i = 0; i < 8; ++i) { a[i] = lo[i]; a[i + 8] = hi[i]; }
            v16h bf = load_b_frag(Vp, SEQ, dt * 16, k0, lane); // B[ks][dd]=Vt[dd][ks]
            acc = wmma_f16(a, bf, acc);
        }
        const int dd = dt * 16 + nlo;
#pragma unroll
        for (int r = 0; r < 8; ++r) {
            const int s = q0base + qt * 16 + mhi + r;
            Out[((size_t)b * SEQ + s) * EMB + h * DH + dd] = acc[r];
        }
    }
}

// ---------------------------------------------------------------------------
extern "C" void kernel_launch(void* const* d_in, const int* in_sizes, int n_in,
                              void* d_out, int out_size, void* d_ws, size_t ws_size,
                              hipStream_t stream) {
    const float* X  = (const float*)d_in[0];
    const float* Wq = (const float*)d_in[1];
    const float* bq = (const float*)d_in[2];
    const float* Wk = (const float*)d_in[3];
    const float* bk = (const float*)d_in[4];
    const float* Wv = (const float*)d_in[5];
    const float* bv = (const float*)d_in[6];
    float* out = (float*)d_out;

    // Workspace layout (all 256B aligned); total ~406 MB
    char* ws = (char*)d_ws;
    size_t oz = 0;
    _Float16* Xh = (_Float16*)(ws + oz); oz += (size_t)65536 * EMB * 2;      // 100,663,296
    _Float16* Wh = (_Float16*)(ws + oz); oz += (size_t)NTOT * EMB * 2;       //   3,538,944
    float*  bcat = (float*)(ws + oz);    oz += (size_t)NTOT * 4;
    oz = (oz + 255) & ~(size_t)255;
    const size_t qkv_bytes = (size_t)SEQ * NHEAD * SEQ * DH * 2;             // 100,663,296
    _Float16* Qb = (_Float16*)(ws + oz); oz += qkv_bytes;
    _Float16* Kb = (_Float16*)(ws + oz); oz += qkv_bytes;
    _Float16* Vt = (_Float16*)(ws + oz); oz += qkv_bytes;

    // 1) X -> f16    (50,331,648 elems, 4 per thread)
    const long long nX = (long long)65536 * EMB;
    attn_cvt_f16_kernel<<<(unsigned)(nX / 4 / 256), 256, 0, stream>>>(X, Xh, nX);

    // 2) pack weights + bias
    const int nW = 3 * EMB * EMB;
    attn_pack_w_kernel<<<(nW + 255) / 256, 256, 0, stream>>>(Wq, Wk, Wv, bq, bk, bv, Wh, bcat);

    // 3) QKV projection: grid (N tiles 2304/256 = 9, M tiles 65536/128 = 512)
    attn_qkv_proj_kernel<<<dim3(NTOT / 256, 65536 / 128), 256, 0, stream>>>(
        Xh, Wh, bcat, Qb, Kb, Vt);

    // 4) fused attention: one block per (b,h,qblock64) = 3072*4
    attn_fused_kernel<<<SEQ * NHEAD * (SEQ / 64), 256, 0, stream>>>(Qb, Kb, Vt, out);
}
